// Conv2d_selfAdapt_27805618274709
// MI455X (gfx1250) — matrix operand order repaired and checked by reading, not compiled, not run on hardware
//
#include <hip/hip_runtime.h>

// ---------------------------------------------------------------------------
// Problem constants (from the reference)
// ---------------------------------------------------------------------------
#define BB   16
#define CC   512
#define HH   48
#define WW   48
#define NPIX (HH * WW)   // 2304
#define OC   8           // conv output channels (padded to 16 for WMMA)

typedef __attribute__((ext_vector_type(16))) _Float16 v16h;
typedef __attribute__((ext_vector_type(8)))  float    v8f;
typedef __attribute__((ext_vector_type(4)))  float    v4f;

// ---------------------------------------------------------------------------
// Stage 1: per-pixel conv3x3(512->8) via WMMA, LayerNorm+gumbel+argmax,
//          and channel-sum CS.  Each wave owns one 1x16 pixel strip.
//   grid = 16 images * 18 blocks, block = 256 threads (8 waves, wave32)
//
// GEMM: M=16 pixels, N=16 (8 outs + 8 zero pad), K = 9 offsets x 512 ch.
// Loop order c0(16) -> di(3) -> dj(3): x halo row staged once per (c0,di)
// in LDS as fp16; A/B fragments are LDS gathers; 144 WMMAs per wave.
// ---------------------------------------------------------------------------
__launch_bounds__(256, 1)
__global__ void cs_sel_stage1(const float* __restrict__ x,
                              const float* __restrict__ mw,     // [8][512][3][3]
                              const float* __restrict__ lnw,    // [8]
                              const float* __restrict__ lnb,    // [8]
                              const float* __restrict__ gum,    // [16][8][48][48]
                              float*       __restrict__ cs,     // [16][2304]
                              unsigned*    __restrict__ sel)    // [16][2304]
{
    __shared__ _Float16 wlds[9 * 16 * 32];   // 9216 B: weights [k][o pad16][kk]
    __shared__ _Float16 xlds[8 * 18 * 32];   // 9216 B: x tile  [wave][col18][kk]
    __shared__ float    llds[8 * 16 * OC];   // 4096 B: logits

    const int tid  = threadIdx.x;
    const int lane = tid & 31;
    const int wave = tid >> 5;
    const int half = lane >> 4;       // 16-lane half of the wave
    const int mrow = lane & 15;       // A row (pixel) / B column (output)

    const int b     = blockIdx.x / 18;
    const int btile = blockIdx.x % 18;
    const int t     = btile * 8 + wave;      // 0..143 strips per image
    const int h     = t / 3;                 // 0..47
    const int w0    = (t % 3) * 16;          // 0,16,32

    _Float16* xw = &xlds[wave * (18 * 32)];

    v8f acc = {};                            // fp32 accumulator (16x16)

    for (int c0 = 0; c0 < CC; c0 += 32) {
        // ---- stage weight chunk: wlds[(k*16+o)*32+kk], zero for o>=8 ----
        for (int j = tid; j < 9 * 16 * 32; j += 256) {
            const int k  = j >> 9;           // /512
            const int r  = j & 511;
            const int o  = r >> 5;
            const int kk = r & 31;
            float wv = 0.0f;
            if (o < OC) wv = mw[(o * CC + (c0 + kk)) * 9 + k];
            wlds[j] = (_Float16)wv;
        }
        __syncthreads();

        for (int di = -1; di <= 1; ++di) {
            // ---- stage x halo row (wave-private): 32 ch x 18 cols, fp16 ----
            const int  hh    = h + di;
            const int  hhc   = (hh < 0) ? 0 : ((hh > HH - 1) ? HH - 1 : hh);
            const bool rowok = (hh >= 0) && (hh < HH);
            for (int i = lane; i < 32 * 18; i += 32) {
                const int  c   = i / 18;
                const int  col = i % 18;
                const int  wwp = w0 - 1 + col;
                const int  wwc = (wwp < 0) ? 0 : ((wwp > WW - 1) ? WW - 1 : wwp);
                const long ga  = (long)(b * CC + c0 + c) * NPIX + hhc * WW + wwc;
                float xv = x[ga];                         // unconditional load
                xv *= (float)(rowok && (wwp >= 0) && (wwp < WW)); // branchless mask
                xw[col * 32 + c] = (_Float16)xv;
                if (c0 + 32 < CC)                          // prefetch next chunk
                    __builtin_prefetch(&x[ga + (long)32 * NPIX], 0, 1);
            }
            __syncthreads();

#pragma unroll
            for (int dj = 0; dj <= 2; ++dj) {
                const int k = (di + 1) * 3 + dj;
                // A fragment (16x32 fp16) from LDS; pairs are adjacent channels
                v16h a;
#pragma unroll
                for (int e = 0; e < 16; ++e) {
                    const int v  = e >> 1;
                    const int kk = ((v >= 4) ? 16 : 0) + half * 8 + (v & 3) * 2 + (e & 1);
                    a[e] = xw[(mrow + dj) * 32 + kk];
                }
                // B fragment (32x16 fp16): kk in table order, matching A
                v16h bm;
#pragma unroll
                for (int e = 0; e < 16; ++e) {
                    bm[e] = wlds[(k * 16 + mrow) * 32 + ((e >= 8) ? 16 : 0) + half * 8 + (e & 7)];
                }
                acc = __builtin_amdgcn_wmma_f32_16x16x32_f16(
                          /*neg_a=*/false, a, /*neg_b=*/false, bm,
                          /*c_mod=*/(short)0, acc, /*reuse_a=*/false, /*reuse_b=*/false);
            }
            __syncthreads();
        }
    }

    // Spill logits (C layout: lane -> N=lane&15, VGPR r -> M=r+8*half)
    if (mrow < OC) {
#pragma unroll
        for (int r = 0; r < 8; ++r) {
            const int mm = r + 8 * half;
            llds[(wave * 16 + mm) * OC + mrow] = acc[r];
        }
    }
    __syncthreads();

    // Channel sum: each lane sums 256 channels of its pixel, fold halves.
    const long pbase = ((long)b * CC + half * 256) * NPIX + h * WW + (w0 + mrow);
    float s = 0.0f;
#pragma unroll 8
    for (int c = 0; c < 256; ++c) s += x[pbase + (long)c * NPIX];
    s += __shfl_xor(s, 16, 32);

    // Lanes 0..15: LayerNorm over 8 channels + gumbel + argmax, emit CS/sel.
    if (half == 0) {
        const int mm  = mrow;
        const int pix = h * WW + w0 + mm;
        float l[8], mu = 0.0f;
#pragma unroll
        for (int o = 0; o < 8; ++o) { l[o] = llds[(wave * 16 + mm) * OC + o]; mu += l[o]; }
        mu *= 0.125f;
        float var = 0.0f;
#pragma unroll
        for (int o = 0; o < 8; ++o) { const float d = l[o] - mu; var += d * d; }
        var *= 0.125f;
        const float rinv = rsqrtf(var + 1e-6f);

        int best = 0; float bestv = -3.0e38f;
#pragma unroll
        for (int o = 0; o < 8; ++o) {
            const float ln = lnw[o] * (l[o] - mu) * rinv + lnb[o];
            const float sc = ln + gum[(long)(b * 8 + o) * NPIX + pix];  // THETA == 1
            if (sc > bestv) { bestv = sc; best = o; }                   // first-max tie
        }
        cs [(long)b * NPIX + pix] = s;
        sel[(long)b * NPIX + pix] = (unsigned)best;
    }
}

// ---------------------------------------------------------------------------
// Stage 2: out[b,c,h,w] = CS[b,h,w] - CS[b, h+di-1, w+dj-1], broadcast over c.
//   4 pixels per thread -> b128 stores.  grid = 36 blocks * 256 threads.
// ---------------------------------------------------------------------------
__launch_bounds__(256, 1)
__global__ void broadcast_stage2(const float*    __restrict__ cs,
                                 const unsigned* __restrict__ sel,
                                 float*          __restrict__ out)
{
    const int tq   = blockIdx.x * blockDim.x + threadIdx.x;   // 0..9215
    const int b    = tq / (NPIX / 4);
    const int pix0 = (tq % (NPIX / 4)) * 4;

    v4f val;
#pragma unroll
    for (int u = 0; u < 4; ++u) {
        const int pix = pix0 + u;
        const int h = pix / WW, w = pix % WW;
        const unsigned a = sel[b * NPIX + pix];
        const int k  = (a < 4) ? (int)a : (int)a + 1;         // skip center
        const int hh = h + k / 3 - 1;
        const int ww = w + k % 3 - 1;
        const int hhc = (hh < 0) ? 0 : ((hh > HH - 1) ? HH - 1 : hh);
        const int wwc = (ww < 0) ? 0 : ((ww > WW - 1) ? WW - 1 : ww);
        float neigh = cs[b * NPIX + hhc * WW + wwc];          // unconditional
        neigh *= (float)((hh >= 0) && (hh < HH) && (ww >= 0) && (ww < WW));
        val[u] = cs[b * NPIX + pix] - neigh;                  // +1 center, -1 sel
    }

    float* obase = out + (long)b * CC * NPIX + pix0;
#pragma unroll 4
    for (int c = 0; c < CC; ++c)
        *(v4f*)(obase + (long)c * NPIX) = val;                // global_store_b128
}

// ---------------------------------------------------------------------------
extern "C" void kernel_launch(void* const* d_in, const int* in_sizes, int n_in,
                              void* d_out, int out_size, void* d_ws, size_t ws_size,
                              hipStream_t stream)
{
    const float* x   = (const float*)d_in[0];
    const float* mw  = (const float*)d_in[1];
    const float* lnw = (const float*)d_in[2];
    const float* lnb = (const float*)d_in[3];
    const float* gum = (const float*)d_in[4];
    // d_in[5] == init_flag, unused by the forward math.

    float*    out = (float*)d_out;
    float*    cs  = (float*)d_ws;                                     // 147456 B
    unsigned* sel = (unsigned*)((char*)d_ws + (size_t)BB * NPIX * 4); // +147456 B
    (void)in_sizes; (void)n_in; (void)out_size; (void)ws_size;

    cs_sel_stage1<<<BB * 18, 256, 0, stream>>>(x, mw, lnw, lnb, gum, cs, sel);
    broadcast_stage2<<<(BB * NPIX) / 4 / 256, 256, 0, stream>>>(cs, sel, out);
}
